// GFF2_Module_77489799954811
// MI455X (gfx1250) — matrix operand order repaired and checked by reading, not compiled, not run on hardware
//
#include <hip/hip_runtime.h>

// ---------------------------------------------------------------------------
// CDNA5 (gfx1250) wave32 WMMA implementation of GFF2 module.
// Matrix math via v_wmma_f32_16x16x32_bf16; attention pass-2 stages K/V tiles
// through LDS with async global->LDS copies (ASYNCcnt) double-buffered.
// ---------------------------------------------------------------------------

typedef __attribute__((ext_vector_type(16))) __bf16 v16bf;
typedef __attribute__((ext_vector_type(8)))  float  v8f;
typedef __attribute__((ext_vector_type(4)))  int    v4i_;

#define DEV static __device__ __forceinline__

union BF16Frag { unsigned u[8]; v16bf v; };

DEV unsigned short f2bf(float f) {
  union { float f; unsigned u; } x; x.f = f;
  unsigned r = x.u + 0x7FFFu + ((x.u >> 16) & 1u);   // round-to-nearest-even
  return (unsigned short)(r >> 16);
}

// A-matrix (16x32 bf16) per-VGPR K base: lanes 0-15 rows, lanes 16-31 rows w/ +8 K offset
DEV int a_kidx(int v, int half) { return (v < 4 ? 2 * v : 16 + 2 * (v - 4)) + 8 * half; }
// B-matrix (32x16 bf16) per-VGPR K base
DEV int b_kidx(int v, int half) { return (v < 4 ? 2 * v : 16 + 2 * (v - 4)) + 8 * half; }

DEV v8f wmma_bf16(const BF16Frag& a, const BF16Frag& b, v8f c) {
  return __builtin_amdgcn_wmma_f32_16x16x32_bf16(
      /*neg_a=*/false, a.v, /*neg_b=*/false, b.v,
      /*c_mod=*/(short)0, c, /*reuse_a=*/false, /*reuse_b=*/false);
}

// ---- CDNA5 async global->LDS copy (ASYNCcnt) with compile-safe fallback ----
#if __has_builtin(__builtin_amdgcn_global_load_async_to_lds_b128)
#define HAVE_ASYNC_LDS 1
DEV void cp_async16(const void* g, void* l) {
  // probe-derived signature: (v4i* global_src, v4i* lds_dst, imm offset, imm cpol)
  __builtin_amdgcn_global_load_async_to_lds_b128((v4i_*)g, (v4i_*)l, 0, 0);
}
#if __has_builtin(__builtin_amdgcn_s_wait_asynccnt)
#define WAIT_ASYNC(n) __builtin_amdgcn_s_wait_asynccnt(n)
#else
#define WAIT_ASYNC(n) asm volatile("s_wait_asynccnt %0" ::"i"(n) : "memory")
#endif
#else
#define HAVE_ASYNC_LDS 0
DEV void cp_async16(const void* g, void* l) {           // synchronous fallback
  *(uint4*)l = *(const uint4*)g;
}
#define WAIT_ASYNC(n)
#endif

// ---------------- problem constants ----------------
#define BB   4
#define CC   512
#define NN_  4096           // H*W = 64*64
#define KD   64             // key dim
#define NU   16384          // Hu*Wu = 128*128
#define CM   128            // Cm = C2/4

// ---------------- prep kernels ----------------
__global__ void prep_wqwk(const float* __restrict__ wq, const float* __restrict__ wk,
                          unsigned short* __restrict__ wqbf, unsigned short* __restrict__ wkbf) {
  int i = blockIdx.x * blockDim.x + threadIdx.x;
  if (i < KD * CC) { wqbf[i] = f2bf(wq[i]); wkbf[i] = f2bf(wk[i]); }
}

__global__ void prep_w2(const float* __restrict__ gw1, unsigned short* __restrict__ w2) {
  int i = blockIdx.x * blockDim.x + threadIdx.x;
  if (i >= CM * CC * 9) return;
  int dx = i % 3, dy = (i / 3) % 3, ci = (i / 9) % CC, co = i / (9 * CC);
  // w2[(dy*3+dx)][co][ci]  (ci contiguous -> B-fragment dword pairs)
  w2[((size_t)((dy * 3 + dx) * CM + co)) * CC + ci] = f2bf(gw1[i]);
}

__global__ void prep_x(const float* __restrict__ x,
                       unsigned short* __restrict__ xbf, unsigned short* __restrict__ xT) {
  size_t i = (size_t)blockIdx.x * blockDim.x + threadIdx.x;
  if (i >= (size_t)BB * CC * NN_) return;
  int tok = (int)(i & (NN_ - 1));
  int c   = (int)((i >> 12) & (CC - 1));
  int b   = (int)(i >> 21);
  unsigned short v = f2bf(x[i]);
  xbf[i] = v;                                        // (B,C,N)  channel-major
  xT[((size_t)(b * NN_ + tok)) * CC + c] = v;        // (B,N,C)  token-major
}

__global__ void prep_xl(const float* __restrict__ xl, unsigned short* __restrict__ xlT) {
  size_t i = (size_t)blockIdx.x * blockDim.x + threadIdx.x;
  if (i >= (size_t)BB * CC * NU) return;
  int p = (int)(i & (NU - 1));
  int c = (int)((i >> 14) & (CC - 1));
  int b = (int)(i >> 23);
  xlT[((size_t)(b * NU + p)) * CC + c] = f2bf(xl[i]); // (B,P,C2) pixel-major
}

// ---------------- Q/K projection: qT/kT = (x^T * W^T) + bias ----------------
__global__ void proj_qk(const unsigned short* __restrict__ xT,
                        const unsigned short* __restrict__ wqbf,
                        const unsigned short* __restrict__ wkbf,
                        const float* __restrict__ bq, const float* __restrict__ bk,
                        unsigned short* __restrict__ qT, unsigned short* __restrict__ kT) {
  const int lane = threadIdx.x & 31, wave = threadIdx.x >> 5;
  const int half = lane >> 4, nn = lane & 15;
  const int b = blockIdx.x >> 6;
  const int tokbase = ((blockIdx.x & 63) << 6) + wave * 16;
  v8f accq[4] = {}; v8f acck[4] = {};
  const unsigned short* xrow = xT + ((size_t)(b * NN_ + tokbase + nn)) * CC;
  for (int ch = 0; ch < 16; ++ch) {
    const int cib = ch * 32;
    BF16Frag a;
#pragma unroll
    for (int v = 0; v < 8; ++v)
      a.u[v] = *(const unsigned*)(xrow + cib + a_kidx(v, half));
#pragma unroll
    for (int t = 0; t < 4; ++t) {
      const int kd = t * 16 + nn;
      BF16Frag fq, fk;
#pragma unroll
      for (int v = 0; v < 8; ++v) {
        const int kk = cib + b_kidx(v, half);
        fq.u[v] = *(const unsigned*)(wqbf + kd * CC + kk);
        fk.u[v] = *(const unsigned*)(wkbf + kd * CC + kk);
      }
      accq[t] = wmma_bf16(a, fq, accq[t]);
      acck[t] = wmma_bf16(a, fk, acck[t]);
    }
  }
#pragma unroll
  for (int t = 0; t < 4; ++t) {
    const int kd = t * 16 + nn;
    const float bqv = bq[kd], bkv = bk[kd];
#pragma unroll
    for (int v = 0; v < 8; ++v) {
      const int tok = tokbase + v + 8 * half;    // D row M = v + 8*half
      qT[((size_t)(b * NN_ + tok)) * KD + kd] = f2bf(accq[t][v] + bqv);
      kT[((size_t)(b * NN_ + tok)) * KD + kd] = f2bf(acck[t][v] + bkv);
    }
  }
}

// ---------------- pass 1: softmax row stats (online max / sumexp) ----------------
__global__ void attn_stats(const unsigned short* __restrict__ qT,
                           const unsigned short* __restrict__ kT,
                           float* __restrict__ mrow, float* __restrict__ lrow) {
  const int lane = threadIdx.x & 31, wave = threadIdx.x >> 5;
  const int half = lane >> 4, nn = lane & 15;
  const int b = blockIdx.x >> 6;
  const int tokbase = ((blockIdx.x & 63) << 6) + wave * 16;
  BF16Frag aq[2];
  {
    const unsigned short* qrow = qT + ((size_t)(b * NN_ + tokbase + nn)) * KD;
#pragma unroll
    for (int d = 0; d < 2; ++d)
#pragma unroll
      for (int v = 0; v < 8; ++v)
        aq[d].u[v] = *(const unsigned*)(qrow + d * 32 + a_kidx(v, half));
  }
  float mr[8], lr[8];
#pragma unroll
  for (int v = 0; v < 8; ++v) { mr[v] = -1e30f; lr[v] = 0.f; }
  for (int jb = 0; jb < NN_; jb += 16) {
    v8f s = {};
    const unsigned short* krow = kT + ((size_t)(b * NN_ + jb + nn)) * KD;
#pragma unroll
    for (int d = 0; d < 2; ++d) {
      BF16Frag bk;
#pragma unroll
      for (int v = 0; v < 8; ++v)
        bk.u[v] = *(const unsigned*)(krow + d * 32 + b_kidx(v, half));
      s = wmma_bf16(aq[d], bk, s);
    }
#pragma unroll
    for (int v = 0; v < 8; ++v) {
      float sv = s[v];
      float nm = fmaxf(mr[v], sv);
      lr[v] = lr[v] * __expf(mr[v] - nm) + __expf(sv - nm);
      mr[v] = nm;
    }
  }
  // reduce across 16 lanes (columns); xor masks <16 stay within each half
#pragma unroll
  for (int v = 0; v < 8; ++v) {
#pragma unroll
    for (int off = 1; off < 16; off <<= 1) {
      float om = __shfl_xor(mr[v], off, 32);
      float ol = __shfl_xor(lr[v], off, 32);
      float nm = fmaxf(mr[v], om);
      lr[v] = lr[v] * __expf(mr[v] - nm) + ol * __expf(om - nm);
      mr[v] = nm;
    }
  }
  if (nn == 0) {
#pragma unroll
    for (int v = 0; v < 8; ++v) {
      const int row = tokbase + v + 8 * half;
      mrow[b * NN_ + row] = mr[v];
      lrow[b * NN_ + row] = lr[v];
    }
  }
}

// ---------------- pass 2: O = softmax(QK^T) * V^T ----------------
// K/V tiles shared by all 4 waves -> double-buffered async LDS staging.
__global__ void attn_out(const unsigned short* __restrict__ qT,
                         const unsigned short* __restrict__ kT,
                         const unsigned short* __restrict__ xbf,
                         const float* __restrict__ mrow, const float* __restrict__ lrow,
                         float* __restrict__ obuf) {
  __shared__ __align__(16) unsigned short kbuf[2][32][64];   // 2 x 4 KB  (keys x d)
  __shared__ __align__(16) unsigned short vbuf[2][128][32];  // 2 x 8 KB  (chan x keys)
  __shared__ unsigned pbuf[4][16][18];                       // per-wave P tile (D->A transpose)
  const int tid  = threadIdx.x;
  const int lane = tid & 31, wave = tid >> 5;
  const int half = lane >> 4, nn = lane & 15;
  const int cblk = blockIdx.x & 3;
  const int qblk = (blockIdx.x >> 2) & 63;
  const int b    = blockIdx.x >> 8;
  const int tokbase = qblk * 64 + wave * 16;
  const int cbase   = cblk * 128;

  const unsigned short* kbase_g = kT  + ((size_t)b * NN_) * KD;
  const unsigned short* vbase_g = xbf + ((size_t)(b * CC + cbase)) * NN_;

  // stage one 32-key K tile + 128x32 V tile into buffer `buf`
  auto stage = [&](int buf, int jb) {
#pragma unroll
    for (int r = 0; r < 2; ++r) {            // K: 256 16B chunks / 128 threads
      const int chunk = tid + 128 * r;
      const int row = chunk >> 3, col = (chunk & 7) * 8;
      cp_async16(kbase_g + (size_t)(jb + row) * KD + col, &kbuf[buf][row][col]);
    }
#pragma unroll
    for (int r = 0; r < 4; ++r) {            // V: 512 16B chunks / 128 threads
      const int chunk = tid + 128 * r;
      const int row = chunk >> 2, col = (chunk & 3) * 8;
      cp_async16(vbase_g + (size_t)row * NN_ + jb + col, &vbuf[buf][row][col]);
    }
  };

  BF16Frag aq[2];
  {
    const unsigned short* qrow = qT + ((size_t)(b * NN_ + tokbase + nn)) * KD;
#pragma unroll
    for (int d = 0; d < 2; ++d)
#pragma unroll
      for (int v = 0; v < 8; ++v)
        aq[d].u[v] = *(const unsigned*)(qrow + d * 32 + a_kidx(v, half));
  }
  float mloc[8];
#pragma unroll
  for (int v = 0; v < 8; ++v) mloc[v] = mrow[b * NN_ + tokbase + v + 8 * half];
  v8f oacc[8] = {};
  unsigned short* prow = (unsigned short*)pbuf[wave];

  stage(0, 0);                               // prologue: first tile in flight
  for (int jb = 0; jb < NN_; jb += 32) {
    const int cur = (jb >> 5) & 1;
    if (jb + 32 < NN_) {
      stage(cur ^ 1, jb + 32);               // prefetch next tile
      WAIT_ASYNC(6);                         // wait for current tile only
    } else {
      WAIT_ASYNC(0);
    }
    __syncthreads();

#pragma unroll
    for (int st = 0; st < 2; ++st) {
      v8f s = {};
#pragma unroll
      for (int d = 0; d < 2; ++d) {
        BF16Frag bk;
#pragma unroll
        for (int v = 0; v < 8; ++v)
          bk.u[v] = *(const unsigned*)(&kbuf[cur][st * 16 + nn][d * 32 + b_kidx(v, half)]);
        s = wmma_bf16(aq[d], bk, s);
      }
#pragma unroll
      for (int v = 0; v < 8; ++v)            // D elem (M=v+8h, N=nn) -> P[row][key]
        prow[(v + 8 * half) * 36 + st * 16 + nn] = f2bf(__expf(s[v] - mloc[v]));
    }
    BF16Frag ap;                             // A-fragment of P (same-wave LDS RAW)
#pragma unroll
    for (int v = 0; v < 8; ++v)
      ap.u[v] = pbuf[wave][nn][a_kidx(v, half) >> 1];
#pragma unroll
    for (int ct = 0; ct < 8; ++ct) {
      BF16Frag bv;
#pragma unroll
      for (int v = 0; v < 8; ++v)
        bv.u[v] = *(const unsigned*)(&vbuf[cur][ct * 16 + nn][b_kidx(v, half)]);
      oacc[ct] = wmma_bf16(ap, bv, oacc[ct]);
    }
    __syncthreads();                         // protect buffers before next stage
  }

  float il[8];
#pragma unroll
  for (int v = 0; v < 8; ++v) il[v] = 1.0f / lrow[b * NN_ + tokbase + v + 8 * half];
#pragma unroll
  for (int ct = 0; ct < 8; ++ct) {
    const int c = cbase + ct * 16 + nn;
#pragma unroll
    for (int v = 0; v < 8; ++v)
      obuf[((size_t)(b * CC + c)) * NN_ + tokbase + v + 8 * half] = oacc[ct][v] * il[v];
  }
}

// ---------------- gamma gate: conv3x3 (9 shifted GEMMs) -> BN -> ReLU -> 1x1 -> sigmoid ----------------
__global__ void gamma_gate(const unsigned short* __restrict__ xlT,
                           const unsigned short* __restrict__ w2,
                           const float* __restrict__ bn_scale, const float* __restrict__ bn_bias,
                           const float* __restrict__ bn_mean,  const float* __restrict__ bn_var,
                           const float* __restrict__ gw2, const float* __restrict__ gb2,
                           float* __restrict__ gamma) {
  const int lane = threadIdx.x & 31, wave = threadIdx.x >> 5;
  const int half = lane >> 4, nn = lane & 15;
  const int xblk = blockIdx.x & 1;
  const int y    = (blockIdx.x >> 1) & 127;
  const int b    = blockIdx.x >> 8;
  const int xbase = xblk * 64 + wave * 16;
  const int px = xbase + nn;                  // A-row pixel column
  v8f hacc[8] = {};
  for (int dydx = 0; dydx < 9; ++dydx) {
    const int dy = dydx / 3 - 1, dx = dydx % 3 - 1;
    const int ys = y + dy;
    if (ys < 0 || ys >= 128) continue;        // wave-uniform
    const int xs = px + dx;
    const bool valid = (xs >= 0) && (xs < 128);
    const unsigned short* arow = xlT + ((size_t)(b * NU + ys * 128 + xs)) * CC;
    for (int ch = 0; ch < 16; ++ch) {
      const int cib = ch * 32;
      BF16Frag a;
#pragma unroll
      for (int v = 0; v < 8; ++v)
        a.u[v] = valid ? *(const unsigned*)(arow + cib + a_kidx(v, half)) : 0u;
#pragma unroll
      for (int ct = 0; ct < 8; ++ct) {
        BF16Frag bw;
        const unsigned short* wrow = w2 + ((size_t)(dydx * CM + ct * 16 + nn)) * CC + cib;
#pragma unroll
        for (int v = 0; v < 8; ++v)
          bw.u[v] = *(const unsigned*)(wrow + b_kidx(v, half));
        hacc[ct] = wmma_bf16(a, bw, hacc[ct]);
      }
    }
  }
  float g[8];
#pragma unroll
  for (int v = 0; v < 8; ++v) g[v] = 0.f;
#pragma unroll
  for (int ct = 0; ct < 8; ++ct) {
    const int co = ct * 16 + nn;
    const float rs = rsqrtf(bn_var[co] + 1e-5f);
    const float sc = bn_scale[co] * rs;
    const float sh = bn_bias[co] - bn_mean[co] * sc;
    const float wv = gw2[co];
#pragma unroll
    for (int v = 0; v < 8; ++v) {
      float h = fmaxf(hacc[ct][v] * sc + sh, 0.f);   // BN + ReLU
      g[v] += h * wv;                                 // 1x1 partial (over lane's co)
    }
  }
#pragma unroll
  for (int v = 0; v < 8; ++v)
#pragma unroll
    for (int off = 1; off < 16; off <<= 1)
      g[v] += __shfl_xor(g[v], off, 32);
  if (nn == 0) {
    const float gb = gb2[0];
#pragma unroll
    for (int v = 0; v < 8; ++v) {
      const int xo = xbase + v + 8 * half;
      const float z = g[v] + gb;
      gamma[b * NU + y * 128 + xo] = 1.0f / (1.0f + __expf(-z));
    }
  }
}

// ---------------- final: bilinear(align_corners) + nearest + gamma blend ----------------
__global__ void final_blend(const float* __restrict__ x, const float* __restrict__ obuf,
                            const float* __restrict__ gamma, float* __restrict__ out) {
  size_t idx = (size_t)blockIdx.x * blockDim.x + threadIdx.x;
  if (idx >= (size_t)BB * CC * NU) return;
  const int xu = (int)(idx & 127);
  const int yu = (int)((idx >> 7) & 127);
  const int c  = (int)((idx >> 14) & (CC - 1));
  const int b  = (int)(idx >> 23);
  const float scale = 63.0f / 127.0f;
  const float sy = yu * scale, sx = xu * scale;
  int y0 = (int)sy, x0 = (int)sx;
  int y1 = y0 + 1 > 63 ? 63 : y0 + 1;
  int x1 = x0 + 1 > 63 ? 63 : x0 + 1;
  const float wy = sy - (float)y0, wx = sx - (float)x0;
  const float* op = obuf + ((size_t)(b * CC + c)) * NN_;
  const float v00 = op[y0 * 64 + x0], v01 = op[y0 * 64 + x1];
  const float v10 = op[y1 * 64 + x0], v11 = op[y1 * 64 + x1];
  const float o_up = (v00 * (1.f - wx) + v01 * wx) * (1.f - wy)
                   + (v10 * (1.f - wx) + v11 * wx) * wy;
  const float xv = x[((size_t)(b * CC + c)) * NN_ + (yu >> 1) * 64 + (xu >> 1)];
  const float g = gamma[b * NU + yu * 128 + xu];
  out[idx] = (1.0f - g) * o_up + g * xv;
}

// ---------------- host launch ----------------
extern "C" void kernel_launch(void* const* d_in, const int* in_sizes, int n_in,
                              void* d_out, int out_size, void* d_ws, size_t ws_size,
                              hipStream_t stream) {
  (void)in_sizes; (void)n_in; (void)out_size; (void)ws_size;
  const float* x        = (const float*)d_in[0];
  const float* xl       = (const float*)d_in[1];
  const float* wq       = (const float*)d_in[2];
  const float* bq       = (const float*)d_in[3];
  const float* wk       = (const float*)d_in[4];
  const float* bk       = (const float*)d_in[5];
  const float* gw1      = (const float*)d_in[6];
  const float* bn_scale = (const float*)d_in[7];
  const float* bn_bias  = (const float*)d_in[8];
  const float* bn_mean  = (const float*)d_in[9];
  const float* bn_var   = (const float*)d_in[10];
  const float* gw2      = (const float*)d_in[11];
  const float* gb2      = (const float*)d_in[12];
  float* out = (float*)d_out;

  // workspace layout (bytes)
  char* ws = (char*)d_ws;
  const size_t OFF_WQBF  = 0;                              //  64*512*2
  const size_t OFF_WKBF  = OFF_WQBF + 65536;
  const size_t OFF_W2    = OFF_WKBF + 65536;               //  9*128*512*2
  const size_t OFF_XBF   = OFF_W2   + 1179648;             //  4*512*4096*2
  const size_t OFF_XT    = OFF_XBF  + 16777216;            //  4*4096*512*2
  const size_t OFF_XLT   = OFF_XT   + 16777216;            //  4*16384*512*2
  const size_t OFF_QT    = OFF_XLT  + 67108864;            //  4*4096*64*2
  const size_t OFF_KT    = OFF_QT   + 2097152;
  const size_t OFF_MROW  = OFF_KT   + 2097152;             //  4*4096*4
  const size_t OFF_LROW  = OFF_MROW + 65536;
  const size_t OFF_OBUF  = OFF_LROW + 65536;               //  4*512*4096*4
  const size_t OFF_GAMMA = OFF_OBUF + 33554432;            //  4*16384*4

  unsigned short* wqbf = (unsigned short*)(ws + OFF_WQBF);
  unsigned short* wkbf = (unsigned short*)(ws + OFF_WKBF);
  unsigned short* w2   = (unsigned short*)(ws + OFF_W2);
  unsigned short* xbf  = (unsigned short*)(ws + OFF_XBF);
  unsigned short* xT   = (unsigned short*)(ws + OFF_XT);
  unsigned short* xlT  = (unsigned short*)(ws + OFF_XLT);
  unsigned short* qT   = (unsigned short*)(ws + OFF_QT);
  unsigned short* kT   = (unsigned short*)(ws + OFF_KT);
  float* mrow  = (float*)(ws + OFF_MROW);
  float* lrow  = (float*)(ws + OFF_LROW);
  float* obuf  = (float*)(ws + OFF_OBUF);
  float* gam   = (float*)(ws + OFF_GAMMA);

  prep_wqwk<<<(KD * CC) / 256, 256, 0, stream>>>(wq, wk, wqbf, wkbf);
  prep_w2<<<(CM * CC * 9) / 256, 256, 0, stream>>>(gw1, w2);
  prep_x<<<(BB * CC * NN_) / 256, 256, 0, stream>>>(x, xbf, xT);
  prep_xl<<<(BB * CC * NU) / 256, 256, 0, stream>>>(xl, xlT);

  proj_qk<<<BB * (NN_ / 64), 128, 0, stream>>>(xT, wqbf, wkbf, bq, bk, qT, kT);
  attn_stats<<<BB * (NN_ / 64), 128, 0, stream>>>(qT, kT, mrow, lrow);
  attn_out<<<BB * (NN_ / 64) * (CC / 128), 128, 0, stream>>>(qT, kT, xbf, mrow, lrow, obuf);
  gamma_gate<<<BB * 128 * 2, 128, 0, stream>>>(xlT, w2, bn_scale, bn_bias, bn_mean, bn_var,
                                               gw2, gb2, gam);
  final_blend<<<(BB * CC * NU) / 256, 256, 0, stream>>>(x, obuf, gam, out);
}
